// PointnetFPModule_60009283059731
// MI455X (gfx1250) — matrix operand order repaired and checked by reading, not compiled, not run on hardware
//
#include <hip/hip_runtime.h>

// ---------------------------------------------------------------------------
// PointNet++ FP module for MI455X (gfx1250, wave32, WMMA, async LDS copies)
//   k1: 3-NN interpolation + concat -> bf16 activation matrix X (B*N x 768)
//   k2: W1/W2 fp32 -> bf16
//   k3: GEMM1 (65536x768x256) bf16 WMMA + bias/BN/ReLU -> bf16 H1
//   k4: GEMM2 (65536x256x256) bf16 WMMA + bias/BN/ReLU -> fp32 out (B,H,N)
// GEMM blocks: 8 waves share one M-tile; the shared 1KB A-tile per K-step is
// staged in LDS by the async copy engine (ASYNCcnt) with double buffering.
// ---------------------------------------------------------------------------

typedef __attribute__((ext_vector_type(16))) __bf16 v16bf;
typedef __attribute__((ext_vector_type(8)))  __bf16 v8bf;
typedef __attribute__((ext_vector_type(8)))  float  v8f;

typedef int v4i_ __attribute__((vector_size(16)));       // matches builtin param
typedef __attribute__((address_space(1))) v4i_ gv4i;     // global int4
typedef __attribute__((address_space(3))) v4i_ lv4i;     // LDS int4

#define BB   16
#define NN   4096
#define SS   1024
#define C1C  256
#define C2C  512
#define CIN  768
#define HH   256

#if defined(__has_builtin)
#  if __has_builtin(__builtin_amdgcn_global_load_async_to_lds_b128)
#    define HAVE_ASYNC_LDS 1
#  endif
#  if __has_builtin(__builtin_amdgcn_s_wait_asynccnt)
#    define HAVE_WAIT_ASYNC_BUILTIN 1
#  endif
#endif
#ifndef HAVE_ASYNC_LDS
#  define HAVE_ASYNC_LDS 0
#endif

#if defined(HAVE_WAIT_ASYNC_BUILTIN)
#  define WAIT_ASYNC(n) __builtin_amdgcn_s_wait_asynccnt(n)
#elif HAVE_ASYNC_LDS
#  define WAIT_ASYNC(n) asm volatile("s_wait_asynccnt " #n ::: "memory")
#else
#  define WAIT_ASYNC(n) ((void)0)   // staging path is synchronous
#endif

__device__ __forceinline__ unsigned short f2bf(float f) {
  union { float f; unsigned u; } x; x.f = f;
  unsigned u = x.u;
  unsigned r = (u + 0x7FFFu + ((u >> 16) & 1u)) >> 16;   // round-to-nearest-even
  return (unsigned short)r;
}

// ---------------------------------------------------------------------------
// Kernel 1: 3-NN interpolation + feature concat, packed as bf16 rows of 768.
// ---------------------------------------------------------------------------
__global__ __launch_bounds__(256) void fp_interp_pack(
    const float* __restrict__ unknown,      // (B, N, 3)
    const float* __restrict__ known,        // (B, S, 3)
    const float* __restrict__ uf,           // (B, C1, N)
    const float* __restrict__ kf,           // (B, C2, S)
    unsigned short* __restrict__ Xbf)       // (B*N, 768) bf16
{
  __shared__ float kx[SS], ky[SS], kz[SS];
  const int b     = blockIdx.x >> 4;
  const int chunk = blockIdx.x & 15;
  const int tid   = threadIdx.x;

  const float* kb = known + (size_t)b * SS * 3;
  for (int i = tid; i < SS; i += 256) {
    kx[i] = kb[i * 3 + 0];
    ky[i] = kb[i * 3 + 1];
    kz[i] = kb[i * 3 + 2];
  }
  __syncthreads();

  const int    n   = chunk * 256 + tid;
  const size_t row = (size_t)b * NN + n;
  const float  ux  = unknown[row * 3 + 0];
  const float  uy  = unknown[row * 3 + 1];
  const float  uz  = unknown[row * 3 + 2];

  float d0 = 3.4e38f, d1 = 3.4e38f, d2 = 3.4e38f;
  int   i0 = 0, i1 = 0, i2 = 0;
  for (int s = 0; s < SS; ++s) {
    const float dx = ux - kx[s], dy = uy - ky[s], dz = uz - kz[s];
    const float d  = dx * dx + dy * dy + dz * dz;
    if (d < d2) {
      if (d < d1) {
        d2 = d1; i2 = i1;
        if (d < d0) { d1 = d0; i1 = i0; d0 = d; i0 = s; }
        else        { d1 = d;  i1 = s; }
      } else        { d2 = d;  i2 = s; }
    }
  }

  float w0 = 1.0f / (d0 + 1e-8f);
  float w1 = 1.0f / (d1 + 1e-8f);
  float w2 = 1.0f / (d2 + 1e-8f);
  const float inv = 1.0f / (w0 + w1 + w2);
  w0 *= inv; w1 *= inv; w2 *= inv;

  unsigned short* Xrow = Xbf + row * CIN;

  const float* ufb = uf + (size_t)b * C1C * NN + n;
  for (int c = 0; c < C1C; ++c)
    Xrow[c] = f2bf(ufb[(size_t)c * NN]);

  const float* kfb = kf + (size_t)b * C2C * SS;
  for (int c = 0; c < C2C; ++c) {
    const float* p = kfb + (size_t)c * SS;
    Xrow[C1C + c] = f2bf(w0 * p[i0] + w1 * p[i1] + w2 * p[i2]);
  }
}

// ---------------------------------------------------------------------------
// Kernel 2: weight conversion fp32 -> bf16
// ---------------------------------------------------------------------------
__global__ __launch_bounds__(256) void fp_cvt_w(
    const float* __restrict__ W1, const float* __restrict__ W2,
    unsigned short* __restrict__ W1b, unsigned short* __restrict__ W2b)
{
  const int i = blockIdx.x * 256 + threadIdx.x;
  if (i < HH * CIN) W1b[i] = f2bf(W1[i]);
  if (i < HH * HH)  W2b[i] = f2bf(W2[i]);
}

// ---------------------------------------------------------------------------
// Stage one 16x32 bf16 A-tile (1KB, row-major, 64B rows) into LDS.
// Issued by wave 0 only: each lane copies two 16B chunks.
// chunk c in [0,64): row = c>>2, byte-in-row = (c&3)*16.
// ---------------------------------------------------------------------------
__device__ __forceinline__ void stage_a_tile(const __bf16* arow_base, int rowBytes,
                                             int k0, unsigned char* dst, int lane) {
#pragma unroll
  for (int t = 0; t < 2; ++t) {
    const int c = lane + t * 32;
    const char* g = (const char*)arow_base + (size_t)(c >> 2) * rowBytes
                    + (size_t)k0 * 2 + (c & 3) * 16;
    unsigned char* l = dst + c * 16;
#if HAVE_ASYNC_LDS
    __builtin_amdgcn_global_load_async_to_lds_b128(
        (gv4i*)(void*)g, (lv4i*)(void*)l, 0, 0);
#else
    *(uint4*)l = *(const uint4*)g;          // vmem load + ds_store_b128 fallback
#endif
  }
}

// ---------------------------------------------------------------------------
// GEMM + bias + BN + ReLU.  Y(M x 256) = X(M x K) * W^T,  M = 65536.
// One wave per 16x16 output tile; 8 waves per block share tileM; 8192 blocks.
// A staged in LDS via async copies (double buffered); B direct from L2.
// Fragment addressing follows the CDNA5 16-bit WMMA VGPR layouts:
//   A (16x32):  row = m0 + (lane&15); halves at bytes half*16 and 32+half*16
//   B (32x16):  col = n0 + (lane&15); halves at k0 + half*8 and k0 + 16 + half*8
//   C/D (f32):  d[r] -> (m0 + 8*half + r, n0 + (lane&15))
// ---------------------------------------------------------------------------
template <int K, bool FINAL>
__global__ __launch_bounds__(256) void fp_gemm_bn_relu(
    const __bf16* __restrict__ X,        // (M, K) bf16, row-major
    const __bf16* __restrict__ W,        // (256, K) bf16, row-major
    const float* __restrict__ bias,
    const float* __restrict__ g,  const float* __restrict__ beta,
    const float* __restrict__ mu, const float* __restrict__ var,
    unsigned short* __restrict__ outBf,  // (M, 256) bf16 when !FINAL
    float* __restrict__ outF)            // (B, 256, N) fp32 when FINAL
{
  __shared__ alignas(16) unsigned char Atile[2][1024];  // double-buffered 16x32 bf16

  const int wv    = threadIdx.x >> 5;
  const int wid   = blockIdx.x * 8 + wv;
  const int tileN = wid & 15;            // 256/16 = 16 column tiles
  const int tileM = wid >> 4;            // identical for all 8 waves of a block
  const int lane  = threadIdx.x & 31;
  const int half  = lane >> 4;
  const int lcol  = lane & 15;
  const int m0 = tileM << 4;
  const int n0 = tileN << 4;

  const __bf16* arow_base = X + (size_t)m0 * K;               // block A rows
  const __bf16* wrow      = W + (size_t)(n0 + lcol) * K;      // per-lane B row

  v8f acc;
#pragma unroll
  for (int i = 0; i < 8; ++i) acc[i] = 0.0f;

  if (wv == 0) stage_a_tile(arow_base, K * 2, 0, Atile[0], lane);

  int idx = 0;
  for (int k0 = 0; k0 < K; k0 += 32, idx ^= 1) {
    if (wv == 0) {
      if (k0 + 32 < K) {
        stage_a_tile(arow_base, K * 2, k0 + 32, Atile[idx ^ 1], lane);
        WAIT_ASYNC(2);                   // the 2 copies for buf[idx] have landed
      } else {
        WAIT_ASYNC(0);
      }
    }
    __syncthreads();                     // publish Atile[idx] to all 8 waves

    if (k0 + 32 < K) __builtin_prefetch(wrow + k0 + 32, 0, 1);

    const unsigned char* ab = Atile[idx];
    const v8bf alo = *(const v8bf*)(ab + lcol * 64 + half * 16);       // ds_load_b128
    const v8bf ahi = *(const v8bf*)(ab + lcol * 64 + 32 + half * 16);
    const v8bf blo = *(const v8bf*)(wrow + k0 + half * 8);
    const v8bf bhi = *(const v8bf*)(wrow + k0 + 16 + half * 8);

    v16bf a, bf;
#pragma unroll
    for (int i = 0; i < 8; ++i) {
      a[i] = alo[i];  a[i + 8] = ahi[i];
      bf[i] = blo[i]; bf[i + 8] = bhi[i];
    }

    acc = __builtin_amdgcn_wmma_f32_16x16x32_bf16(
        /*neg_a=*/false, a, /*neg_b=*/false, bf,
        /*c_mod=*/(short)0, acc, /*reuse_a=*/false, /*reuse_b=*/false);

    __syncthreads();                     // reads of Atile[idx] done before reuse
  }

  // fused bias + inference BN + ReLU on output channel o = n0 + lcol
  const int   o     = n0 + lcol;
  const float scale = g[o] * rsqrtf(var[o] + 1e-5f);
  const float shift = beta[o] - mu[o] * scale + bias[o] * scale;

#pragma unroll
  for (int r = 0; r < 8; ++r) {
    const int   m = m0 + half * 8 + r;
    float v = fmaxf(acc[r] * scale + shift, 0.0f);
    if (FINAL) {
      const int b = m >> 12;             // m = b*4096 + n; per-lane stores are
      const int n = m & 4095;            // 8 consecutive floats -> b128 merged
      outF[((size_t)b * HH + o) * NN + n] = v;
    } else {
      outBf[(size_t)m * HH + o] = f2bf(v);
    }
  }
}

// ---------------------------------------------------------------------------
extern "C" void kernel_launch(void* const* d_in, const int* in_sizes, int n_in,
                              void* d_out, int out_size, void* d_ws, size_t ws_size,
                              hipStream_t stream) {
  (void)in_sizes; (void)n_in; (void)out_size; (void)ws_size;

  const float* unknown = (const float*)d_in[0];
  const float* known   = (const float*)d_in[1];
  const float* uf      = (const float*)d_in[2];
  const float* kf      = (const float*)d_in[3];
  const float* W1      = (const float*)d_in[4];
  const float* b1      = (const float*)d_in[5];
  const float* g1      = (const float*)d_in[6];
  const float* beta1   = (const float*)d_in[7];
  const float* mu1     = (const float*)d_in[8];
  const float* var1    = (const float*)d_in[9];
  const float* W2      = (const float*)d_in[10];
  const float* b2      = (const float*)d_in[11];
  const float* g2      = (const float*)d_in[12];
  const float* beta2   = (const float*)d_in[13];
  const float* mu2     = (const float*)d_in[14];
  const float* var2    = (const float*)d_in[15];

  char* ws = (char*)d_ws;
  // workspace layout (bytes):
  //   X   : 65536*768*2 = 100663296
  //   H1  : 65536*256*2 =  33554432   @ 100663296
  //   W1b : 256*768*2   =    393216   @ 134217728
  //   W2b : 256*256*2   =    131072   @ 134610944
  unsigned short* Xbf = (unsigned short*)(ws);
  unsigned short* H1  = (unsigned short*)(ws + 100663296ull);
  unsigned short* W1b = (unsigned short*)(ws + 134217728ull);
  unsigned short* W2b = (unsigned short*)(ws + 134610944ull);

  fp_interp_pack<<<BB * (NN / 256), 256, 0, stream>>>(unknown, known, uf, kf, Xbf);
  fp_cvt_w<<<(HH * CIN) / 256, 256, 0, stream>>>(W1, W2, W1b, W2b);

  // 65536/16 M-tiles * 16 N-tiles = 65536 wave-tiles, 8 waves/block -> 8192 blocks
  fp_gemm_bn_relu<CIN, false><<<8192, 256, 0, stream>>>(
      (const __bf16*)Xbf, (const __bf16*)W1b, b1, g1, beta1, mu1, var1,
      H1, nullptr);
  fp_gemm_bn_relu<HH, true><<<8192, 256, 0, stream>>>(
      (const __bf16*)H1, (const __bf16*)W2b, b2, g2, beta2, mu2, var2,
      nullptr, (float*)d_out);
}
